// CustomDeformableDetrMLPPredictionHead_14302241096164
// MI455X (gfx1250) — matrix operand order repaired and checked by reading, not compiled, not run on hardware
//
#include <hip/hip_runtime.h>

// ---------------------------------------------------------------------------
// Types for CDNA5 WMMA
// ---------------------------------------------------------------------------
typedef __attribute__((ext_vector_type(16))) __bf16 v16bf;
typedef __attribute__((ext_vector_type(8)))  float  v8f;

union Frag {
    v16bf v;
    uint4 q[2];
};

// float -> bf16 (round-to-nearest-even), bf16 -> float, via bit ops
static __device__ __forceinline__ unsigned short f2bf(float f) {
    unsigned u = __float_as_uint(f);
    u = (u + 0x7FFFu + ((u >> 16) & 1u)) >> 16;
    return (unsigned short)u;
}
static __device__ __forceinline__ float bf2f(unsigned short v) {
    return __uint_as_float(((unsigned)v) << 16);
}

#define LL 6
#define BB 2
#define QN 300
#define QP 304      // padded to multiple of 16
#define DD 256

// ---------------------------------------------------------------------------
// Kernel A: projections for 7 logical layers (6 x Wq/Wk + Wsub/Wobj on hs[-1])
// Emits bf16 Q_all/K_all (padded rows zeroed) + fp32 gq/gk.
// One block = one (l, b, q) row, 256 threads (one output feature each).
// ---------------------------------------------------------------------------
__global__ void __launch_bounds__(256) proj_kernel(
    const float* __restrict__ hs,   const float* __restrict__ Wq,
    const float* __restrict__ bq,   const float* __restrict__ Wk,
    const float* __restrict__ bk,   const float* __restrict__ Wsub,
    const float* __restrict__ bsub, const float* __restrict__ Wobj,
    const float* __restrict__ bobj, const float* __restrict__ Wg,
    unsigned short* __restrict__ Qv, unsigned short* __restrict__ Kv,
    float* __restrict__ gq, float* __restrict__ gk)
{
    const int q = blockIdx.x;   // 0..303
    const int b = blockIdx.y;   // 0..1
    const int l = blockIdx.z;   // 0..6
    const int t = threadIdx.x;  // 0..255

    __shared__ float hrow[DD];
    __shared__ float redQ[DD];
    __shared__ float redK[DD];

    const int outRow = (l * BB + b) * QP + q;

    if (q >= QN) {   // padded rows: zero everything (block-uniform branch)
        Qv[outRow * DD + t] = 0;
        Kv[outRow * DD + t] = 0;
        if (t == 0) { gq[outRow] = 0.0f; gk[outRow] = 0.0f; }
        return;
    }

    const int srcl = (l == 6) ? 5 : l;
    hrow[t] = hs[((srcl * BB + b) * QN + q) * DD + t];
    __syncthreads();

    const float* WQl = (l == 6) ? Wsub : (Wq + l * DD * DD);
    const float* WKl = (l == 6) ? Wobj : (Wk + l * DD * DD);
    float aq = (l == 6) ? bsub[t] : bq[l * DD + t];
    float ak = (l == 6) ? bobj[t] : bk[l * DD + t];

#pragma unroll 4
    for (int d = 0; d < DD; ++d) {
        const float h = hrow[d];
        aq = fmaf(h, WQl[d * DD + t], aq);
        ak = fmaf(h, WKl[d * DD + t], ak);
    }

    Qv[outRow * DD + t] = f2bf(aq);
    Kv[outRow * DD + t] = f2bf(ak);

    // gate logits: gq = Q . Wg[:256], gk = K . Wg[256:]
    redQ[t] = aq * Wg[t];
    redK[t] = ak * Wg[DD + t];
    __syncthreads();
    for (int s = 128; s > 0; s >>= 1) {
        if (t < s) { redQ[t] += redQ[t + s]; redK[t] += redK[t + s]; }
        __syncthreads();
    }
    if (t == 0) { gq[outRow] = redQ[0]; gk[outRow] = redK[0]; }
}

// ---------------------------------------------------------------------------
// Kernel B: repack W1 (512x256) and W2 (256x256) into bf16 WMMA-B-fragment
// order: flat[(kc*16 + nt)*32*16 + lane*16 + h] = W[K,N],
// with N = nt*16 + lane%16, K = kc*32 + h + 16*(lane/16)   (contiguous k, same order as the A loads).
// Each lane then loads its 16 halves as two contiguous b128s.
// ---------------------------------------------------------------------------
__global__ void __launch_bounds__(256) wprep_kernel(
    const float* __restrict__ W1, const float* __restrict__ W2,
    unsigned short* __restrict__ W1f, unsigned short* __restrict__ W2f)
{
    const int idx = blockIdx.x * blockDim.x + threadIdx.x; // 0..196607
    if (idx < 131072) {
        const int h    = idx & 15;
        const int lane = (idx >> 4) & 31;
        const int nt   = (idx >> 9) & 15;
        const int kc   = idx >> 13;          // 0..15
        const int N = nt * 16 + (lane & 15);
        const int K = kc * 32 + h + ((lane >> 4) << 4);
        W1f[idx] = f2bf(W1[K * 256 + N]);
    } else {
        const int i2   = idx - 131072;       // 0..65535
        const int h    = i2 & 15;
        const int lane = (i2 >> 4) & 31;
        const int nt   = (i2 >> 9) & 15;
        const int kc   = i2 >> 13;           // 0..7
        const int N = nt * 16 + (lane & 15);
        const int K = kc * 32 + h + ((lane >> 4) << 4);
        W2f[i2] = f2bf(W2[K * 256 + N]);
    }
}

// ---------------------------------------------------------------------------
// Kernel C: fused gate + gated-feature build + 3-layer MLP + l-broadcast store.
// Block = 64 threads (2 waves). Wave w handles i = blockIdx.y*2 + w and the
// 16 j's of blockIdx.x: its 16 MLP rows (one per j) share one Q_all row.
// Layer1: 16x512 @ 512x256 via 16 nt x 16 kc WMMA bf16 steps.
// Layer2: 16x256 @ 256x256 via 16 nt x  8 kc WMMA bf16 steps.
// Layer3: per-lane dot with W3 + shuffle butterfly reduction.
// ---------------------------------------------------------------------------
__global__ void __launch_bounds__(64) fused_kernel(
    const unsigned short* __restrict__ Qv, const unsigned short* __restrict__ Kv,
    const float* __restrict__ gq, const float* __restrict__ gk,
    const unsigned short* __restrict__ W1f, const unsigned short* __restrict__ W2f,
    const float* __restrict__ b1, const float* __restrict__ b2,
    const float* __restrict__ W3, const float* __restrict__ b3,
    const float* __restrict__ bg, float* __restrict__ out)
{
    __shared__ __align__(16) unsigned short gated[2][16][512]; // 32 KB
    __shared__ __align__(16) unsigned short x1s[2][16][256];   // 16 KB
    __shared__ float gsh[2][7][16];                            // ~0.9 KB

    const int jt   = blockIdx.x;            // 0..18
    const int b    = blockIdx.z;            // 0..1
    const int w    = threadIdx.x >> 5;      // wave 0/1
    const int lane = threadIdx.x & 31;
    const int i    = blockIdx.y * 2 + w;    // 0..303
    const int nlo  = lane & 15;
    const int khid = lane >> 4;             // 0/1 (upper half-wave)

    // ---- phase 0: gates g[l][j_local] = sigmoid(gq + gk + bg) ----
    const float bgv = bg[0];
    for (int idx = lane; idx < 7 * 16; idx += 32) {
        const int l  = idx >> 4;
        const int jl = idx & 15;
        const float x = gq[(l * BB + b) * QP + i]
                      + gk[(l * BB + b) * QP + jt * 16 + jl] + bgv;
        gsh[w][l][jl] = 1.0f / (1.0f + __expf(-x));
    }
    __syncthreads();

    // ---- phase 1: build gated rows (16 x 512 bf16) in LDS ----
    const int lstride = BB * QP * DD;
    for (int it = 0; it < 256; ++it) {
        const int idx = it * 32 + lane;
        const int row = idx >> 9;       // j_local (uniform per it)
        const int k   = idx & 511;      // contiguous across lanes
        float acc = 0.0f;
        if (k < 256) {
            const int base = (b * QP + i) * DD + k;
#pragma unroll
            for (int l = 0; l < 7; ++l)
                acc = fmaf(gsh[w][l][row], bf2f(Qv[base + l * lstride]), acc);
        } else {
            const int j = jt * 16 + row;
            const int base = (b * QP + j) * DD + (k - 256);
#pragma unroll
            for (int l = 0; l < 7; ++l)
                acc = fmaf(gsh[w][l][row], bf2f(Kv[base + l * lstride]), acc);
        }
        gated[w][row][k] = f2bf(acc);
    }
    __syncthreads();

    // ---- phase 2: layer 1 (512 -> 256), ReLU, bf16 to LDS ----
    for (int nt = 0; nt < 16; ++nt) {
        const float bias = b1[nt * 16 + nlo];
        v8f acc;
#pragma unroll
        for (int v = 0; v < 8; ++v) acc[v] = bias;
        for (int kc = 0; kc < 16; ++kc) {
            Frag a, bb;
            const uint4* ap =
                (const uint4*)&gated[w][nlo][kc * 32 + (khid ? 16 : 0)];
            a.q[0] = ap[0];            // halfs 0-7  : K = base .. base+7
            a.q[1] = ap[1];            // halfs 8-15 : K = base+8 .. base+15
            const uint4* bp = (const uint4*)&W1f[((kc * 16 + nt) * 32 + lane) * 16];
            bb.q[0] = bp[0];
            bb.q[1] = bp[1];
            acc = __builtin_amdgcn_wmma_f32_16x16x32_bf16(
                false, a.v, false, bb.v, (short)0, acc, false, false);
        }
#pragma unroll
        for (int v = 0; v < 8; ++v) {
            const float r = acc[v] > 0.0f ? acc[v] : 0.0f;
            x1s[w][v + (khid << 3)][nt * 16 + nlo] = f2bf(r);
        }
    }
    __syncthreads();

    // ---- phase 3: layer 2 (256 -> 256) + ReLU + layer 3 dot with W3 ----
    float predAcc[8];
#pragma unroll
    for (int v = 0; v < 8; ++v) predAcc[v] = 0.0f;

    for (int nt = 0; nt < 16; ++nt) {
        const float bias = b2[nt * 16 + nlo];
        v8f acc;
#pragma unroll
        for (int v = 0; v < 8; ++v) acc[v] = bias;
        for (int kc = 0; kc < 8; ++kc) {
            Frag a, bb;
            const uint4* ap =
                (const uint4*)&x1s[w][nlo][kc * 32 + (khid ? 16 : 0)];
            a.q[0] = ap[0];
            a.q[1] = ap[1];
            const uint4* bp = (const uint4*)&W2f[((kc * 16 + nt) * 32 + lane) * 16];
            bb.q[0] = bp[0];
            bb.q[1] = bp[1];
            acc = __builtin_amdgcn_wmma_f32_16x16x32_bf16(
                false, a.v, false, bb.v, (short)0, acc, false, false);
        }
        const float w3v = W3[nt * 16 + nlo];
#pragma unroll
        for (int v = 0; v < 8; ++v) {
            const float r = acc[v] > 0.0f ? acc[v] : 0.0f;
            predAcc[v] = fmaf(r, w3v, predAcc[v]);
        }
    }

    // butterfly reduce over the 16 lanes of each half-wave (C cols N=0..15)
#pragma unroll
    for (int m = 1; m < 16; m <<= 1)
#pragma unroll
        for (int v = 0; v < 8; ++v)
            predAcc[v] += __shfl_xor(predAcc[v], m, 16);

    // ---- store: pred broadcast over l = 0..5 ----
    if (nlo == 0 && i < QN) {
        const float b3v = b3[0];
#pragma unroll
        for (int v = 0; v < 8; ++v) {
            const int jl = v + (khid << 3);
            const int j  = jt * 16 + jl;
            if (j < QN) {
                const float p = predAcc[v] + b3v;
                for (int l = 0; l < LL; ++l)
                    out[((l * BB + b) * QN + i) * QN + j] = p;
            }
        }
    }
}

// ---------------------------------------------------------------------------
// Host launcher
// ---------------------------------------------------------------------------
extern "C" void kernel_launch(void* const* d_in, const int* in_sizes, int n_in,
                              void* d_out, int out_size, void* d_ws, size_t ws_size,
                              hipStream_t stream) {
    const float* hs   = (const float*)d_in[0];
    const float* Wq   = (const float*)d_in[1];
    const float* bq   = (const float*)d_in[2];
    const float* Wk   = (const float*)d_in[3];
    const float* bk   = (const float*)d_in[4];
    const float* Wsub = (const float*)d_in[5];
    const float* bsub = (const float*)d_in[6];
    const float* Wobj = (const float*)d_in[7];
    const float* bobj = (const float*)d_in[8];
    const float* Wg   = (const float*)d_in[9];
    const float* bg   = (const float*)d_in[10];
    const float* W1   = (const float*)d_in[11];
    const float* b1   = (const float*)d_in[12];
    const float* W2   = (const float*)d_in[13];
    const float* b2   = (const float*)d_in[14];
    const float* W3   = (const float*)d_in[15];
    const float* b3   = (const float*)d_in[16];
    float* out = (float*)d_out;

    // workspace layout (bytes): Qv | Kv | gq | gk | W1f | W2f  (~4.6 MB total)
    const size_t nQK = (size_t)7 * BB * QP * DD;      // 1,089,536 bf16 elems
    const size_t nG  = (size_t)7 * BB * QP;           // 4,256 floats
    char* ws = (char*)d_ws;
    unsigned short* Qv  = (unsigned short*)ws;
    unsigned short* Kv  = Qv + nQK;
    float*          gqp = (float*)(Kv + nQK);
    float*          gkp = gqp + nG;
    unsigned short* W1f = (unsigned short*)(gkp + nG); // 131072 elems
    unsigned short* W2f = W1f + 131072;                // 65536 elems

    proj_kernel<<<dim3(QP, BB, 7), 256, 0, stream>>>(
        hs, Wq, bq, Wk, bk, Wsub, bsub, Wobj, bobj, Wg, Qv, Kv, gqp, gkp);

    wprep_kernel<<<dim3((131072 + 65536) / 256), 256, 0, stream>>>(
        W1, W2, W1f, W2f);

    fused_kernel<<<dim3(QP / 16, QP / 2, BB), 64, 0, stream>>>(
        Qv, Kv, gqp, gkp, W1f, W2f, b1, b2, W3, b3, bg, out);
}
